// GPTLanguageModel_23579370455386
// MI455X (gfx1250) — matrix operand order repaired and checked by reading, not compiled, not run on hardware
//
#include <hip/hip_runtime.h>
#include <hip/hip_bf16.h>

typedef _Float16 v16h __attribute__((ext_vector_type(16)));
typedef _Float16 v8h  __attribute__((ext_vector_type(8)));
typedef _Float16 v4h  __attribute__((ext_vector_type(4)));
typedef float    v8f  __attribute__((ext_vector_type(8)));

static constexpr int Bsz = 4, Tseq = 512, Dm = 1024, Hn = 16, HDm = 64,
                     Lly = 12, FFn = 4096, Vocab = 50257;
static constexpr int Mrow = Bsz * Tseq;          // 2048
static constexpr int BM = 128, BN = 128, BK = 32;
static constexpr int LPAD = 16;                  // row stride 48 halves = 96B (16B aligned)
static constexpr int LSTR = BK + LPAD;

// ---------------------------------------------------------------------------
// WMMA GEMM: C[M,N] = act(A[M,K](f16) * B[K,N] [+ bias]) [+ resid]
//   Caller guarantees: M % 128 == 0, K % 32 == 0, all buffers < 2GB (int idx).
//   TB:      B element type (float converted on commit, or _Float16)
//   CT:      C store type (float or _Float16)
//   BTRANS:  B stored [N,K] row-major (used for Q*K^T)
//   NGUARD:  N bounds checks (LM head N=50257, attn-out N=64)
//   HASBIAS/HASRES/RELU: compile-time epilogue selection
// Register-buffered pipeline over double-buffered LDS:
//   loadRegs(k+1) issues pure global loads before the 8 WMMAs of step k, so
//   the s_wait_loadcnt lands one full K-step later, hidden behind matrix work.
// ---------------------------------------------------------------------------
template <typename TB, typename CT, bool BTRANS, bool NGUARD,
          bool HASBIAS, bool HASRES, bool RELU>
__global__ __launch_bounds__(256)
void gemm_wmma_kernel(const _Float16* __restrict__ A, const TB* __restrict__ Bp,
                      const float* __restrict__ bias, const float* __restrict__ resid,
                      CT* __restrict__ C,
                      int M, int N, int K, int lda, int ldb, int ldc,
                      int sA1, int sA2, int sB1, int sB2, int sC1, int sC2,
                      int batchH)
{
    __shared__ _Float16 As[2][BM][LSTR];   // row-major: K contiguous
    __shared__ _Float16 Bs[2][BN][LSTR];   // N-major:  K contiguous per column

    const int z  = blockIdx.z;
    const int z1 = z / batchH, z2 = z % batchH;
    A  += z1 * sA1 + z2 * sA2;
    Bp += z1 * sB1 + z2 * sB2;
    C  += z1 * sC1 + z2 * sC2;

    const int tid   = threadIdx.x;
    const int lane  = tid & 31;
    const int wid   = tid >> 5;     // 8 waves
    const int waveM = wid >> 2;     // 0..1  -> 64 rows
    const int waveN = wid & 3;      // 0..3  -> 32 cols

    const int m0 = blockIdx.y * BM;
    const int n0 = blockIdx.x * BN;

    v8f acc[4][2];
    v8f vzero = {};
#pragma unroll
    for (int i = 0; i < 4; ++i)
#pragma unroll
        for (int j = 0; j < 2; ++j) acc[i][j] = vzero;

    // A staging: thread -> (row, 16-half segment)
    const int aRow  = tid >> 1;
    const int aColH = (tid & 1) * 16;
    const _Float16* aBase = A + (m0 + aRow) * lda + aColH;
    // B staging: thread -> one N column, 16 K values
    const int bN = tid & 127;
    const int bK = (tid >> 7) * 16;
    const int gn = n0 + bN;
    const bool bOk = !NGUARD || (gn < N);

    // staging registers (global -> regs, pure loads)
    uint4 ra0, ra1;
    uint4 rb0, rb1;          // BTRANS path (f16 source)
    float rbf[16];           // weight path (f32 source)

    auto loadRegs = [&](int k0) {
        const _Float16* ag = aBase + k0;
        ra0 = ((const uint4*)ag)[0];
        ra1 = ((const uint4*)ag)[1];
        if (BTRANS) {
            if (bOk) {
                const _Float16* bg = (const _Float16*)Bp + gn * ldb + (k0 + bK);
                rb0 = ((const uint4*)bg)[0];
                rb1 = ((const uint4*)bg)[1];
            } else {
                rb0 = make_uint4(0, 0, 0, 0);
                rb1 = make_uint4(0, 0, 0, 0);
            }
        } else {
            if (bOk) {
                const TB* bg = Bp + (k0 + bK) * ldb + gn;
#pragma unroll
                for (int kk = 0; kk < 16; ++kk) rbf[kk] = (float)bg[kk * ldb];
            } else {
#pragma unroll
                for (int kk = 0; kk < 16; ++kk) rbf[kk] = 0.f;
            }
        }
    };

    auto commit = [&](int buf) {
        ((uint4*)&As[buf][aRow][aColH])[0] = ra0;
        ((uint4*)&As[buf][aRow][aColH])[1] = ra1;
        if (BTRANS) {
            ((uint4*)&Bs[buf][bN][bK])[0] = rb0;
            ((uint4*)&Bs[buf][bN][bK])[1] = rb1;
        } else {
            _Float16 tmp[16];
#pragma unroll
            for (int kk = 0; kk < 16; ++kk) tmp[kk] = (_Float16)rbf[kk];
            ((uint4*)&Bs[buf][bN][bK])[0] = ((uint4*)tmp)[0];
            ((uint4*)&Bs[buf][bN][bK])[1] = ((uint4*)tmp)[1];
        }
    };

    const int nk = K / BK;
    loadRegs(0);

    const int arow = lane & 15;
    const int akb  = (lane >> 4) * 8;    // A: lanes 0-15 K0-7/16-23, 16-31 K8-15/24-31
    const int bcol = lane & 15;
    const int bkb  = (lane >> 4) * 8;    // B: lanes 0-15 K0-7/16-23, 16-31 K8-15/24-31

    for (int kt = 0; kt < nk; ++kt) {
        const int cur = kt & 1;
        commit(cur);
        __syncthreads();
        if (kt + 1 < nk) loadRegs((kt + 1) * BK);

        v16h afr[4];
#pragma unroll
        for (int i = 0; i < 4; ++i) {
            const int r = waveM * 64 + i * 16 + arow;
            const v8h lo = *(const v8h*)&As[cur][r][akb];
            const v8h hi = *(const v8h*)&As[cur][r][16 + akb];
            afr[i] = __builtin_shufflevector(lo, hi, 0,1,2,3,4,5,6,7,8,9,10,11,12,13,14,15);
        }
        v16h bfr[2];
#pragma unroll
        for (int j = 0; j < 2; ++j) {
            const int cc = waveN * 32 + j * 16 + bcol;
            const v8h b0 = *(const v8h*)&Bs[cur][cc][bkb];
            const v8h b1 = *(const v8h*)&Bs[cur][cc][bkb + 16];
            bfr[j] = __builtin_shufflevector(b0, b1, 0,1,2,3,4,5,6,7,8,9,10,11,12,13,14,15);
        }

#pragma unroll
        for (int i = 0; i < 4; ++i)
#pragma unroll
            for (int j = 0; j < 2; ++j)
                acc[i][j] = __builtin_amdgcn_wmma_f32_16x16x32_f16(
                    false, afr[i], false, bfr[j], (short)0, acc[i][j], false, false);
    }

    // ---- epilogue: straight-line, coalesced (16 lanes = 16 consecutive n)
    const int nIn  = lane & 15;
    const int mOff = (lane >> 4) * 8;
#pragma unroll
    for (int j = 0; j < 2; ++j) {
        const int n = n0 + waveN * 32 + j * 16 + nIn;
        if (NGUARD && n >= N) continue;
        float bb = 0.f;
        if (HASBIAS) bb = bias[n];
#pragma unroll
        for (int i = 0; i < 4; ++i) {
            const int mBase = m0 + waveM * 64 + i * 16 + mOff;
#pragma unroll
            for (int r = 0; r < 8; ++r) {
                float val = acc[i][j][r];
                if (HASBIAS) val += bb;
                if (RELU && val < 0.f) val = 0.f;
                const int off = (mBase + r) * ldc + n;
                if (HASRES) val += resid[off];
                C[off] = (CT)val;
            }
        }
    }
}

// ---------------------------------------------------------------------------
// LayerNorm: f32 row in -> f16 row out. D == 1024, 256 threads, float4 I/O.
// ---------------------------------------------------------------------------
__global__ __launch_bounds__(256)
void layernorm_f16_kernel(const float* __restrict__ x, const float* __restrict__ g,
                          const float* __restrict__ bta, _Float16* __restrict__ out)
{
    __shared__ float red[256];
    const int row = blockIdx.x;
    const int tid = threadIdx.x;
    const int base = row * Dm;
    const float4 v = *(const float4*)&x[base + tid * 4];
    red[tid] = v.x + v.y + v.z + v.w; __syncthreads();
    for (int st = 128; st > 0; st >>= 1) { if (tid < st) red[tid] += red[tid + st]; __syncthreads(); }
    const float mean = red[0] * (1.f / Dm);
    __syncthreads();
    const float dx = v.x - mean, dy = v.y - mean, dz = v.z - mean, dw = v.w - mean;
    red[tid] = dx * dx + dy * dy + dz * dz + dw * dw; __syncthreads();
    for (int st = 128; st > 0; st >>= 1) { if (tid < st) red[tid] += red[tid + st]; __syncthreads(); }
    const float rstd = rsqrtf(red[0] * (1.f / Dm) + 1e-5f);
    const float4 gg = *(const float4*)&g[tid * 4];
    const float4 bb = *(const float4*)&bta[tid * 4];
    v4h o;
    o[0] = (_Float16)(dx * rstd * gg.x + bb.x);
    o[1] = (_Float16)(dy * rstd * gg.y + bb.y);
    o[2] = (_Float16)(dz * rstd * gg.z + bb.z);
    o[3] = (_Float16)(dw * rstd * gg.w + bb.w);
    *(v4h*)&out[base + tid * 4] = o;
}

// ---------------------------------------------------------------------------
// Causal softmax, T=512 cols; scale 1/8; f32 scores -> f16 probs (packed b32).
// ---------------------------------------------------------------------------
__global__ __launch_bounds__(256)
void softmax_causal_kernel(const float* __restrict__ S, _Float16* __restrict__ P)
{
    __shared__ float red[256];
    const int r   = blockIdx.x;
    const int i   = r & (Tseq - 1);
    const int tid = threadIdx.x;
    const int base = r * Tseq;
    const float2 sv = *(const float2*)&S[base + tid * 2];
    float v0 = sv.x * 0.125f;
    float v1 = sv.y * 0.125f;
    if (tid * 2 > i)     v0 = -3.0e38f;
    if (tid * 2 + 1 > i) v1 = -3.0e38f;
    red[tid] = fmaxf(v0, v1); __syncthreads();
    for (int st = 128; st > 0; st >>= 1) { if (tid < st) red[tid] = fmaxf(red[tid], red[tid + st]); __syncthreads(); }
    const float mx = red[0]; __syncthreads();
    const float e0 = __expf(v0 - mx), e1 = __expf(v1 - mx);
    red[tid] = e0 + e1; __syncthreads();
    for (int st = 128; st > 0; st >>= 1) { if (tid < st) red[tid] += red[tid + st]; __syncthreads(); }
    const float inv = 1.f / red[0];
    _Float16 p2[2] = { (_Float16)(e0 * inv), (_Float16)(e1 * inv) };
    *(unsigned int*)&P[base + tid * 2] = *(unsigned int*)p2;
}

// ---------------------------------------------------------------------------
// Embedding (float4): x[b,t,:] = tok_emb[idx[b,t],:] + pos_emb[t,:]
// ---------------------------------------------------------------------------
__global__ __launch_bounds__(256)
void embed_kernel(const int* __restrict__ idx, const float* __restrict__ tok,
                  const float* __restrict__ pos, float* __restrict__ x)
{
    const int i4 = blockIdx.x * 256 + threadIdx.x;      // one float4 each
    if (i4 >= Mrow * (Dm / 4)) return;
    const int bt = i4 / (Dm / 4);
    const int d4 = i4 % (Dm / 4);
    const int t  = bt & (Tseq - 1);
    const float4 tv = *(const float4*)&tok[idx[bt] * Dm + d4 * 4];
    const float4 pv = *(const float4*)&pos[t * Dm + d4 * 4];
    float4 o; o.x = tv.x + pv.x; o.y = tv.y + pv.y; o.z = tv.z + pv.z; o.w = tv.w + pv.w;
    *(float4*)&x[i4 * 4] = o;
}

// ---------------------------------------------------------------------------
extern "C" void kernel_launch(void* const* d_in, const int* in_sizes, int n_in,
                              void* d_out, int out_size, void* d_ws, size_t ws_size,
                              hipStream_t stream)
{
    (void)in_sizes; (void)n_in; (void)out_size; (void)ws_size;
    const int*   idx  = (const int*)  d_in[0];
    const float* tok  = (const float*)d_in[1];
    const float* pos  = (const float*)d_in[2];
    const float* Wq   = (const float*)d_in[3];
    const float* Wk   = (const float*)d_in[4];
    const float* Wv   = (const float*)d_in[5];
    const float* Wo   = (const float*)d_in[6];
    const float* bo   = (const float*)d_in[7];
    const float* ln1g = (const float*)d_in[8];
    const float* ln1b = (const float*)d_in[9];
    const float* ln2g = (const float*)d_in[10];
    const float* ln2b = (const float*)d_in[11];
    const float* W1   = (const float*)d_in[12];
    const float* b1   = (const float*)d_in[13];
    const float* W2   = (const float*)d_in[14];
    const float* b2   = (const float*)d_in[15];
    const float* lnfg = (const float*)d_in[16];
    const float* lnfb = (const float*)d_in[17];
    const float* Wlm  = (const float*)d_in[18];
    const float* blm  = (const float*)d_in[19];
    float* out = (float*)d_out;

    size_t off = 0;
    auto alloc = [&](size_t bytes) -> void* {
        off = (off + 255) & ~(size_t)255;
        void* p = (char*)d_ws + off;
        off += bytes;
        return p;
    };
    float*     x      = (float*)    alloc((size_t)Mrow * Dm * 4);
    _Float16*  h      = (_Float16*) alloc((size_t)Mrow * Dm * 2);
    _Float16*  q      = (_Float16*) alloc((size_t)Mrow * Dm * 2);
    _Float16*  k      = (_Float16*) alloc((size_t)Mrow * Dm * 2);
    _Float16*  v      = (_Float16*) alloc((size_t)Mrow * Dm * 2);
    _Float16*  o      = (_Float16*) alloc((size_t)Mrow * Dm * 2);
    float*     scores = (float*)    alloc((size_t)Bsz * Hn * Tseq * Tseq * 4);
    _Float16*  P      = (_Float16*) alloc((size_t)Bsz * Hn * Tseq * Tseq * 2);
    _Float16*  ffh    = (_Float16*) alloc((size_t)Mrow * FFn * 2);

    const dim3 blk(256);

    // 1) embedding
    embed_kernel<<<dim3(Mrow * (Dm / 4) / 256), blk, 0, stream>>>(idx, tok, pos, x);

    for (int l = 0; l < Lly; ++l) {
        const float* Wq_l = Wq + (size_t)l * Dm * Dm;
        const float* Wk_l = Wk + (size_t)l * Dm * Dm;
        const float* Wv_l = Wv + (size_t)l * Dm * Dm;
        const float* Wo_l = Wo + (size_t)l * Dm * Dm;
        const float* W1_l = W1 + (size_t)l * Dm * FFn;
        const float* W2_l = W2 + (size_t)l * FFn * Dm;

        // 2) LN1
        layernorm_f16_kernel<<<dim3(Mrow), blk, 0, stream>>>(x, ln1g + l * Dm, ln1b + l * Dm, h);

        // 3) q, k, v = h @ W{q,k,v}   (f16 out)
        const dim3 gQKV(Dm / BN, Mrow / BM, 1);
        gemm_wmma_kernel<float, _Float16, false, false, false, false, false>
            <<<gQKV, blk, 0, stream>>>(h, Wq_l, nullptr, nullptr, q,
                Mrow, Dm, Dm, Dm, Dm, Dm, 0, 0, 0, 0, 0, 0, 1);
        gemm_wmma_kernel<float, _Float16, false, false, false, false, false>
            <<<gQKV, blk, 0, stream>>>(h, Wk_l, nullptr, nullptr, k,
                Mrow, Dm, Dm, Dm, Dm, Dm, 0, 0, 0, 0, 0, 0, 1);
        gemm_wmma_kernel<float, _Float16, false, false, false, false, false>
            <<<gQKV, blk, 0, stream>>>(h, Wv_l, nullptr, nullptr, v,
                Mrow, Dm, Dm, Dm, Dm, Dm, 0, 0, 0, 0, 0, 0, 1);

        // 4) scores[b,h] = Q_bh (T x 64) @ K_bh^T (64 x T)
        {
            const dim3 g(Tseq / BN, Tseq / BM, Bsz * Hn);
            gemm_wmma_kernel<_Float16, float, true, false, false, false, false>
                <<<g, blk, 0, stream>>>(q, k, nullptr, nullptr, scores,
                    Tseq, Tseq, HDm, Dm, Dm, Tseq,
                    Tseq * Dm, HDm, Tseq * Dm, HDm,
                    Hn * Tseq * Tseq, Tseq * Tseq, Hn);
        }

        // 5) causal softmax -> P (f16)
        softmax_causal_kernel<<<dim3(Bsz * Hn * Tseq), blk, 0, stream>>>(scores, P);

        // 6) o[b, :, h*64..] = P_bh (T x T) @ V_bh (T x 64)   (N=64, guarded)
        {
            const dim3 g(1, Tseq / BM, Bsz * Hn);
            gemm_wmma_kernel<_Float16, _Float16, false, true, false, false, false>
                <<<g, blk, 0, stream>>>(P, v, nullptr, nullptr, o,
                    Tseq, HDm, Tseq, Tseq, Dm, Dm,
                    Hn * Tseq * Tseq, Tseq * Tseq,
                    Tseq * Dm, HDm, Tseq * Dm, HDm, Hn);
        }

        // 7) x = x + o @ Wo + bo
        gemm_wmma_kernel<float, float, false, false, true, true, false>
            <<<gQKV, blk, 0, stream>>>(o, Wo_l, bo + l * Dm, x, x,
                Mrow, Dm, Dm, Dm, Dm, Dm, 0, 0, 0, 0, 0, 0, 1);

        // 8) LN2
        layernorm_f16_kernel<<<dim3(Mrow), blk, 0, stream>>>(x, ln2g + l * Dm, ln2b + l * Dm, h);

        // 9) ffh = relu(h @ W1 + b1)   (f16 out)
        {
            const dim3 g(FFn / BN, Mrow / BM, 1);
            gemm_wmma_kernel<float, _Float16, false, false, true, false, true>
                <<<g, blk, 0, stream>>>(h, W1_l, b1 + l * FFn, nullptr, ffh,
                    Mrow, FFn, Dm, Dm, FFn, FFn, 0, 0, 0, 0, 0, 0, 1);
        }

        // 10) x = x + ffh @ W2 + b2
        gemm_wmma_kernel<float, float, false, false, true, true, false>
            <<<gQKV, blk, 0, stream>>>(ffh, W2_l, b2 + l * Dm, x, x,
                Mrow, Dm, FFn, FFn, Dm, Dm, 0, 0, 0, 0, 0, 0, 1);
    }

    // 11) final LN + LM head (N=50257, guarded)
    layernorm_f16_kernel<<<dim3(Mrow), blk, 0, stream>>>(x, lnfg, lnfb, h);
    {
        const dim3 g((Vocab + BN - 1) / BN, Mrow / BM, 1);
        gemm_wmma_kernel<float, float, false, true, true, false, false>
            <<<g, blk, 0, stream>>>(h, Wlm, blm, nullptr, out,
                Mrow, Vocab, Dm, Dm, Vocab, Vocab, 0, 0, 0, 0, 0, 0, 1);
    }
}
